// PathwayAwareGNN_54400055771427
// MI455X (gfx1250) — matrix operand order repaired and checked by reading, not compiled, not run on hardware
//
#include <hip/hip_runtime.h>
#include <hip/hip_bf16.h>

typedef __attribute__((ext_vector_type(16))) __bf16 v16bf;
typedef __attribute__((ext_vector_type(8)))  __bf16 v8bf;
typedef __attribute__((ext_vector_type(8)))  float  v8f;

#define HID 256   // hidden width (fixed by problem; WMMA layout depends on it)

// ---------------------------------------------------------------------------
// K0: deg[i]=1.0 (self loop), colsum[c]=0
__global__ void k_init(float* __restrict__ deg, float* __restrict__ colsum, int Nn) {
    int i = blockIdx.x * blockDim.x + threadIdx.x;
    if (i < Nn)  deg[i] = 1.0f;
    if (i < HID) colsum[i] = 0.0f;
}

// K1: degree scatter over dst
__global__ void k_deg(const long long* __restrict__ ei, float* __restrict__ deg, int E) {
    int e = blockIdx.x * blockDim.x + threadIdx.x;
    if (e < E) {
        long long d = ei[(long long)E + e];
        unsafeAtomicAdd(&deg[d], 1.0f);
    }
}

// K2: dis = rsqrt(deg) (in place), s init = dis^2 * x  (self-loop term of layer 1)
__global__ void k_dis_s(const float* __restrict__ x, float* __restrict__ deg_dis,
                        float* __restrict__ s, int Nn) {
    int i = blockIdx.x * blockDim.x + threadIdx.x;
    if (i < Nn) {
        float r = rsqrtf(deg_dis[i]);   // deg >= 1 always (self loop)
        deg_dis[i] = r;
        s[i] = r * r * x[i];
    }
}

// K3: per-edge norm + scalar layer-1 scatter: s[dst] += norm * x[src]
__global__ void k_norm_s(const long long* __restrict__ ei, const float* __restrict__ dis,
                         const float* __restrict__ x, float* __restrict__ nrm,
                         float* __restrict__ s, int E) {
    int e = blockIdx.x * blockDim.x + threadIdx.x;
    if (e < E) {
        long long sN = ei[e];
        long long dN = ei[(long long)E + e];
        float nv = dis[sN] * dis[dN];
        nrm[e] = nv;
        unsafeAtomicAdd(&s[dN], nv * x[sN]);
    }
}

// Kswz: pre-swizzle W2 (f32 row-major [K=256][N=256]) into the exact WMMA
// B-operand lane layout, bf16. Flat index ((kt*16+nt)*32+lane)*16 + j holds
// W2[kt*32 + (j<8 ? (lane>>4)*8 + j : 16 + (lane>>4)*8 + (j-8))][nt*16 + (lane&15)].
__global__ void k_w2_swizzle(const float* __restrict__ W2, __bf16* __restrict__ w2s) {
    int o = blockIdx.x * blockDim.x + threadIdx.x;   // 0 .. 65535
    int j    = o & 15;
    int lane = (o >> 4) & 31;
    int ntk  = o >> 9;
    int nt   = ntk & 15;
    int kt   = ntk >> 4;
    int k = kt * 32 + (j < 8 ? (lane >> 4) * 8 + j : 16 + (lane >> 4) * 8 + (j - 8));
    int n = nt * 16 + (lane & 15);
    w2s[o] = (__bf16)W2[k * HID + n];
}

// K5a: agg self-loop init: agg[i,c] = dis[i]^2 * relu(s[i]*W1[c] + b1[c])
__global__ void k_agg_init(const float* __restrict__ dis, const float* __restrict__ s,
                           const float* __restrict__ W1, const float* __restrict__ b1,
                           float* __restrict__ agg, int Nn) {
    int i = blockIdx.x;      // node
    int c = threadIdx.x;     // channel (256)
    if (i < Nn) {
        float r  = dis[i];
        float sv = s[i];
        float h  = fmaxf(fmaf(sv, W1[c], b1[c]), 0.0f);
        agg[(size_t)i * HID + c] = r * r * h;
    }
}

// K5b: 256-wide edge scatter. One wave per edge (grid-stride); each lane owns
// 8 contiguous channels; h1[src,c] recomputed from scalar s[src] (4B/edge read
// instead of 1KB). Atomic f32 adds land in L2 (agg = 51MB << 192MB L2).
__global__ void k_scatter(const long long* __restrict__ ei, const float* __restrict__ nrm,
                          const float* __restrict__ s, const float* __restrict__ W1,
                          const float* __restrict__ b1, float* __restrict__ agg, int E) {
    int lane = threadIdx.x & 31;
    int wid  = (blockIdx.x * blockDim.x + threadIdx.x) >> 5;
    int nw   = (gridDim.x * blockDim.x) >> 5;
    int c0   = lane * 8;
    float w1r[8], b1r[8];
#pragma unroll
    for (int u = 0; u < 8; ++u) { w1r[u] = W1[c0 + u]; b1r[u] = b1[c0 + u]; }
    for (int e = wid; e < E; e += nw) {
        long long sN = ei[e];
        long long dN = ei[(long long)E + e];
        float nv = nrm[e];
        float sv = s[sN];
        float* outp = agg + (size_t)dN * HID + c0;
#pragma unroll
        for (int u = 0; u < 8; ++u) {
            float h = fmaxf(fmaf(sv, w1r[u], b1r[u]), 0.0f);
            unsafeAtomicAdd(outp + u, nv * h);
        }
    }
}

// K6: f32 -> bf16 convert for the WMMA A operand
__global__ void k_cvt(const float* __restrict__ agg, __bf16* __restrict__ aggb, int n) {
    int i = blockIdx.x * blockDim.x + threadIdx.x;
    if (i < n) aggb[i] = (__bf16)agg[i];
}

// K7: h2 = relu(agg @ W2 + b2) via v_wmma_f32_16x16x32_bf16, fused column-sum.
// One wave per 16x16 output tile; K-loop of 8 WMMAs. h2 is never stored.
__global__ void k_gemm_wmma(const __bf16* __restrict__ aggb, const __bf16* __restrict__ w2s,
                            const float* __restrict__ b2, float* __restrict__ colsum,
                            int Nn, int Mt) {
    int wave = threadIdx.x >> 5;
    int lane = threadIdx.x & 31;
    int t = blockIdx.x * (blockDim.x >> 5) + wave;
    if (t >= Mt * 16) return;                 // wave-uniform: EXEC all-1 for WMMA
    int mt = t >> 4;
    int nt = t & 15;
    int half = lane >> 4;
    int l16  = lane & 15;
    int mrow  = mt * 16 + l16;
    int mload = mrow < Nn ? mrow : Nn - 1;    // clamp (N=50000 is 16-aligned anyway)
    // A layout (16-bit, 16x32 MxK): lane holds row m=lane%16; j=0..7 -> K=h*8+j,
    // j=8..15 -> K=16+h*8+(j-8): two contiguous 16B chunks of the row-major row.
    const __bf16* arow  = aggb + (size_t)mload * HID + half * 8;
    const __bf16* bbase = w2s + ((size_t)nt * 32 + lane) * 16;  // +kt*8192 per step
    v8f acc = {};
#pragma unroll
    for (int kt = 0; kt < 8; ++kt) {
        v8bf alo = *(const v8bf*)(arow + kt * 32);
        v8bf ahi = *(const v8bf*)(arow + kt * 32 + 16);
        v16bf a = __builtin_shufflevector(alo, ahi, 0,1,2,3,4,5,6,7,8,9,10,11,12,13,14,15);
        v16bf b = *(const v16bf*)(bbase + (size_t)kt * (16 * 32 * 16));
        acc = __builtin_amdgcn_wmma_f32_16x16x32_bf16(false, a, false, b,
                                                      (short)0, acc, false, false);
    }
    // Epilogue: bias + relu + column partial sum. C/D layout: VGPR r, lane L:
    // row = mt*16 + r + 8*(L>>4), col = nt*16 + (L&15).
    int n = nt * 16 + l16;
    float bn = b2[n];
    float tot = 0.0f;
#pragma unroll
    for (int r = 0; r < 8; ++r) {
        int mr = mt * 16 + r + 8 * half;
        float v = fmaxf(acc[r] + bn, 0.0f);
        if (mr < Nn) tot += v;
    }
    tot += __shfl_xor(tot, 16, 32);           // combine the two half-lanes (same col)
    if (half == 0) unsafeAtomicAdd(&colsum[n], tot);
}

// K8: head. pooled = (colsum/N) @ Wp + bp; z = relu(pooled@Wc1+bc1); out = z@Wc2+bc2
__global__ void k_head(const float* __restrict__ colsum,
                       const float* __restrict__ Wp,  const float* __restrict__ bp,
                       const float* __restrict__ Wc1, const float* __restrict__ bc1,
                       const float* __restrict__ Wc2, const float* __restrict__ bc2,
                       float* __restrict__ out, int Nn, int P, int Q, int NC) {
    __shared__ float ps[128];
    __shared__ float zs[128];
    int tid = threadIdx.x;
    float invN = 1.0f / (float)Nn;
    if (tid < P) {
        float a = bp[tid];
        for (int k = 0; k < HID; ++k) a = fmaf(colsum[k] * invN, Wp[k * P + tid], a);
        ps[tid] = a;
    }
    __syncthreads();
    if (tid < Q) {
        float z = bc1[tid];
        for (int p = 0; p < P; ++p) z = fmaf(ps[p], Wc1[p * Q + tid], z);
        zs[tid] = fmaxf(z, 0.0f);
    }
    __syncthreads();
    if (tid < NC) {
        float o = bc2[tid];
        for (int q = 0; q < Q; ++q) o = fmaf(zs[q], Wc2[q * NC + tid], o);
        out[tid] = o;
    }
}

// ---------------------------------------------------------------------------
extern "C" void kernel_launch(void* const* d_in, const int* in_sizes, int n_in,
                              void* d_out, int out_size, void* d_ws, size_t ws_size,
                              hipStream_t stream) {
    const float*     x   = (const float*)d_in[0];
    const long long* ei  = (const long long*)d_in[1];   // int64 edge_index [2,E]
    // d_in[2] (gene_to_pathway_map) is unused by the reference.
    const float* W1  = (const float*)d_in[3];
    const float* b1  = (const float*)d_in[4];
    const float* W2  = (const float*)d_in[5];
    const float* b2  = (const float*)d_in[6];
    const float* Wp  = (const float*)d_in[7];
    const float* bp  = (const float*)d_in[8];
    const float* Wc1 = (const float*)d_in[9];
    const float* bc1 = (const float*)d_in[10];
    const float* Wc2 = (const float*)d_in[11];
    const float* bc2 = (const float*)d_in[12];

    const int Nn = in_sizes[0];          // 50000
    const int E  = in_sizes[1] / 2;      // 1600000
    const int P  = in_sizes[8];          // 100
    const int Q  = in_sizes[10];         // 128
    const int NC = in_sizes[12];         // 5

    // Workspace carve-up (~84 MB total)
    char* w = (char*)d_ws;
    auto aup = [](size_t v) { return (v + 255) & ~(size_t)255; };
    float*  deg    = (float*)w;  w += aup((size_t)Nn * 4);        // deg -> dis (in place)
    float*  s      = (float*)w;  w += aup((size_t)Nn * 4);
    float*  nrm    = (float*)w;  w += aup((size_t)E * 4);
    float*  agg    = (float*)w;  w += aup((size_t)Nn * HID * 4);
    __bf16* aggb   = (__bf16*)w; w += aup((size_t)Nn * HID * 2);
    __bf16* w2s    = (__bf16*)w; w += aup((size_t)HID * HID * 2);
    float*  colsum = (float*)w;  w += aup((size_t)HID * 4);
    (void)ws_size; (void)n_in; (void)out_size;

    const int nbN = (Nn + 255) / 256;
    const int nbE = (E + 255) / 256;

    k_init      <<<nbN, 256, 0, stream>>>(deg, colsum, Nn);
    k_deg       <<<nbE, 256, 0, stream>>>(ei, deg, E);
    k_dis_s     <<<nbN, 256, 0, stream>>>(x, deg, s, Nn);
    k_norm_s    <<<nbE, 256, 0, stream>>>(ei, deg, x, nrm, s, E);
    k_w2_swizzle<<<(HID * HID + 255) / 256, 256, 0, stream>>>(W2, w2s);
    k_agg_init  <<<Nn, 256, 0, stream>>>(deg, s, W1, b1, agg, Nn);
    k_scatter   <<<6400, 256, 0, stream>>>(ei, nrm, s, W1, b1, agg, E);
    k_cvt       <<<(Nn * HID + 255) / 256, 256, 0, stream>>>(agg, aggb, Nn * HID);
    const int Mt = (Nn + 15) / 16;
    k_gemm_wmma <<<(Mt * 16 + 7) / 8, 256, 0, stream>>>(aggb, w2s, b2, colsum, Nn, Mt);
    k_head      <<<1, 128, 0, stream>>>(colsum, Wp, bp, Wc1, bc1, Wc2, bc2,
                                        (float*)d_out, Nn, P, Q, NC);
}
